// AttentionBlock_46806553592473
// MI455X (gfx1250) — matrix operand order repaired and checked by reading, not compiled, not run on hardware
//
#include <hip/hip_runtime.h>
#include <hip/hip_bf16.h>

typedef __attribute__((ext_vector_type(16))) __bf16 v16bf;
typedef __attribute__((ext_vector_type(8)))  __bf16 v8bf;
typedef __attribute__((ext_vector_type(8)))  float  v8f;

#define B_    8
#define C_    512
#define N_    1024
#define HEADS 8
#define HD    64
#define GROUPS 32
#define EPS_  1e-5f

// ---------------------------------------------------------------------------
// CDNA5 WMMA operand loaders (wave32).
// A (16x32 bf16, MxK): lane l holds row M=l&15; K chunks [hi*8, hi*8+8) and
// [16+hi*8, 16+hi*8+8)  (hi = l>>4).  -> two 16B loads.
// B (32x16 bf16, KxN): lane l holds col N=l&15; K chunks [hi*8, hi*8+8) and
// [16+hi*8, 16+hi*8+8)  (same as A). -> two 16B loads.
// C/D (16x16 f32): VGPR j, lane l -> row = j + 8*hi, col = l&15.
// ---------------------------------------------------------------------------
__device__ __forceinline__ v16bf load_a16(const __bf16* rowp, int hi) {
  v8bf a0 = *(const v8bf*)(rowp + hi * 8);
  v8bf a1 = *(const v8bf*)(rowp + 16 + hi * 8);
  return __builtin_shufflevector(a0, a1, 0, 1, 2, 3, 4, 5, 6, 7,
                                          8, 9, 10, 11, 12, 13, 14, 15);
}
__device__ __forceinline__ v16bf load_b16(const __bf16* colp, int hi) {
  v8bf b0 = *(const v8bf*)(colp + hi * 8);
  v8bf b1 = *(const v8bf*)(colp + 16 + hi * 8);
  return __builtin_shufflevector(b0, b1, 0, 1, 2, 3, 4, 5, 6, 7,
                                          8, 9, 10, 11, 12, 13, 14, 15);
}
__device__ __forceinline__ v8f wmma_bf16(v16bf a, v16bf b, v8f c) {
  return __builtin_amdgcn_wmma_f32_16x16x32_bf16(false, a, false, b,
                                                 (short)0, c, false, false);
}

// ---------------------------------------------------------------------------
// fp32 -> bf16 elementwise convert (weights)
// ---------------------------------------------------------------------------
__global__ void cvt_f32_bf16(const float* __restrict__ src,
                             __bf16* __restrict__ dst, int n) {
  for (int i = blockIdx.x * blockDim.x + threadIdx.x; i < n;
       i += gridDim.x * blockDim.x)
    dst[i] = (__bf16)src[i];
}

// ---------------------------------------------------------------------------
// GroupNorm: one block per (b, group).  Writes h transposed -> [B][N][C] bf16
// so downstream WMMA B-operands are contiguous.
// ---------------------------------------------------------------------------
__global__ __launch_bounds__(256) void gn_kernel(
    const float* __restrict__ x, const float* __restrict__ w,
    const float* __restrict__ bias, __bf16* __restrict__ h_t) {
  const int CPG = C_ / GROUPS;  // 16
  int b = blockIdx.x / GROUPS, g = blockIdx.x % GROUPS;
  int t = threadIdx.x;
  const float* xg = x + ((size_t)b * C_ + g * CPG) * N_;

  float s = 0.f, s2 = 0.f;
  for (int e = t; e < CPG * N_; e += 256) {
    float v = xg[e];
    s += v;
    s2 += v * v;
  }
  __shared__ float r1[256], r2[256];
  r1[t] = s;
  r2[t] = s2;
  __syncthreads();
  for (int off = 128; off > 0; off >>= 1) {
    if (t < off) { r1[t] += r1[t + off]; r2[t] += r2[t + off]; }
    __syncthreads();
  }
  const float inv_n = 1.f / (float)(CPG * N_);
  float mean = r1[0] * inv_n;
  float var  = r2[0] * inv_n - mean * mean;
  float rstd = rsqrtf(var + EPS_);

  for (int e = t; e < CPG * N_; e += 256) {
    int cl = e >> 10, n = e & (N_ - 1);
    int c = g * CPG + cl;
    float v = (xg[e] - mean) * rstd * w[c] + bias[c];
    h_t[((size_t)b * N_ + n) * C_ + c] = (__bf16)v;
  }
}

// ---------------------------------------------------------------------------
// QKV GEMM: qkv[o,n] = Wqkv[o, :] . h[:, n]  per batch.
// One wave -> 16(o) x 128(n) tile (8 accumulators), K = 512.  A (weights)
// loaded once per K-step and reused across 8 WMMAs.
// q,k stored transposed [b][h][n][d]; v stored [b][h][d][n].
// ---------------------------------------------------------------------------
__global__ __launch_bounds__(256) void qkv_gemm(
    const __bf16* __restrict__ wq, const float* __restrict__ qbias,
    const __bf16* __restrict__ h_t, __bf16* __restrict__ qT,
    __bf16* __restrict__ kT, __bf16* __restrict__ vM) {
  const int OT = (3 * C_) / 16;  // 96
  const int NG = N_ / 128;       // 8
  int wid = blockIdx.x * 8 + (threadIdx.x >> 5);
  int lane = threadIdx.x & 31, lo = lane & 15, hi = lane >> 4;
  int b = wid / (OT * NG);
  int r = wid % (OT * NG);
  int ot = r / NG, ng = r % NG;
  int o0 = ot * 16, n0 = ng * 128;

  v8f acc[8] = {};
  const __bf16* arow = wq + (size_t)(o0 + lo) * C_;
  for (int kc = 0; kc < C_ / 32; ++kc) {
    v16bf a = load_a16(arow + kc * 32, hi);
#pragma unroll
    for (int j = 0; j < 8; ++j) {
      const __bf16* bcol =
          h_t + ((size_t)b * N_ + n0 + j * 16 + lo) * C_ + kc * 32;
      acc[j] = wmma_bf16(a, load_b16(bcol, hi), acc[j]);
    }
  }

  int which = o0 / C_;  // 0=q 1=k 2=v (uniform per wave)
  int oc0 = o0 % C_;
  int hh = oc0 / HD;    // uniform per wave
  int d0 = oc0 % HD;
  size_t bh = (size_t)b * HEADS + hh;
#pragma unroll
  for (int jv = 0; jv < 8; ++jv) {
    int d = d0 + jv + 8 * hi;
    float bv = qbias[o0 + jv + 8 * hi];
#pragma unroll
    for (int j = 0; j < 8; ++j) {
      int n = n0 + j * 16 + lo;
      float val = acc[j][jv] + bv;
      if (which == 0)      qT[(bh * N_ + n) * HD + d] = (__bf16)val;
      else if (which == 1) kT[(bh * N_ + n) * HD + d] = (__bf16)val;
      else                 vM[(bh * HD + d) * N_ + n] = (__bf16)val;
    }
  }
}

// ---------------------------------------------------------------------------
// Fused attention: one 4-wave block per (b, head, 16 query rows).
// Phase1: scores = scale * q^T k via WMMA (m-tiles split across waves) -> LDS.
// Phase2: softmax per row (shfl reductions); un-normalized exp stored bf16,
//         1/sum folded into output.
// Phase3: O^T = P . v^T via WMMA (one 16-wide d-tile per wave); output
//         transposed to [B][N][C] bf16 for the projection GEMM.
// ---------------------------------------------------------------------------
#define SC_STRIDE 1025   // f32, +1 to rotate banks
#define P_STRIDE  1048   // bf16, conflict-free b128 row reads

__global__ __launch_bounds__(128) void attn_kernel(
    const __bf16* __restrict__ qT, const __bf16* __restrict__ kT,
    const __bf16* __restrict__ vM, __bf16* __restrict__ oT) {
  const float scale = 0.125f;  // 1/sqrt(64)
  __shared__ float s_sc[16 * SC_STRIDE];
  __shared__ __attribute__((aligned(32))) __bf16 s_p[16 * P_STRIDE];
  __shared__ float s_den[16];

  int blk = blockIdx.x;
  int nb = blk & 63, hh = (blk >> 6) & 7, b = blk >> 9;
  int n0 = nb * 16;
  int wave = threadIdx.x >> 5;
  int lane = threadIdx.x & 31, lo = lane & 15, hi = lane >> 4;
  size_t bh = (size_t)b * HEADS + hh;

  // Phase 1: scores
  const __bf16* qrow = qT + (bh * N_ + n0 + lo) * HD;
  v16bf a0 = load_a16(qrow, hi);
  v16bf a1 = load_a16(qrow + 32, hi);
  for (int mt = wave; mt < N_ / 16; mt += 4) {
    int m0 = mt * 16;
    const __bf16* krow = kT + (bh * N_ + m0 + lo) * HD;
    v8f sc = {};
    sc = wmma_bf16(a0, load_b16(krow, hi), sc);
    sc = wmma_bf16(a1, load_b16(krow + 32, hi), sc);
#pragma unroll
    for (int jv = 0; jv < 8; ++jv)
      s_sc[(jv + 8 * hi) * SC_STRIDE + m0 + lo] = sc[jv] * scale;
  }
  __syncthreads();

  // Phase 2: softmax (wave w owns rows 4w..4w+3)
  for (int rr = wave * 4; rr < wave * 4 + 4; ++rr) {
    float mx = -3.4e38f;
    for (int m = lane; m < N_; m += 32)
      mx = fmaxf(mx, s_sc[rr * SC_STRIDE + m]);
#pragma unroll
    for (int off = 16; off > 0; off >>= 1)
      mx = fmaxf(mx, __shfl_xor(mx, off, 32));
    float sum = 0.f;
    for (int m = lane; m < N_; m += 32) {
      float e = __expf(s_sc[rr * SC_STRIDE + m] - mx);
      s_p[rr * P_STRIDE + m] = (__bf16)e;
      sum += e;
    }
#pragma unroll
    for (int off = 16; off > 0; off >>= 1) sum += __shfl_xor(sum, off, 32);
    if (lane == 0) s_den[rr] = 1.f / sum;
  }
  __syncthreads();

  // Phase 3: O accumulation; wave owns d-tile = wave
  int dt = wave;
  v8f acc = {};
  for (int mc = 0; mc < N_ / 32; ++mc) {
    const __bf16* prow = &s_p[lo * P_STRIDE + mc * 32];
    v16bf pa = load_a16(prow, hi);
    const __bf16* vcol = vM + (bh * HD + dt * 16 + lo) * N_ + mc * 32;
    acc = wmma_bf16(pa, load_b16(vcol, hi), acc);
  }
#pragma unroll
  for (int jv = 0; jv < 8; ++jv) {
    int rrow = jv + 8 * hi;  // local n
    float val = acc[jv] * s_den[rrow];
    oT[((size_t)b * N_ + n0 + rrow) * C_ + hh * HD + dt * 16 + lo] =
        (__bf16)val;
  }
}

// ---------------------------------------------------------------------------
// Projection GEMM + bias + residual, fp32 output [B][C][N].
// One wave -> 16(o) x 128(n) tile (8 accumulators), K = 512.
// ---------------------------------------------------------------------------
__global__ __launch_bounds__(256) void proj_gemm(
    const __bf16* __restrict__ wp, const float* __restrict__ pbias,
    const __bf16* __restrict__ o_t, const float* __restrict__ x,
    float* __restrict__ out) {
  const int OT = C_ / 16;   // 32
  const int NG = N_ / 128;  // 8
  int wid = blockIdx.x * 8 + (threadIdx.x >> 5);
  int lane = threadIdx.x & 31, lo = lane & 15, hi = lane >> 4;
  int b = wid / (OT * NG);
  int r = wid % (OT * NG);
  int ot = r / NG, ng = r % NG;
  int o0 = ot * 16, n0 = ng * 128;

  v8f acc[8] = {};
  const __bf16* arow = wp + (size_t)(o0 + lo) * C_;
  for (int kc = 0; kc < C_ / 32; ++kc) {
    v16bf a = load_a16(arow + kc * 32, hi);
#pragma unroll
    for (int j = 0; j < 8; ++j) {
      const __bf16* bcol =
          o_t + ((size_t)b * N_ + n0 + j * 16 + lo) * C_ + kc * 32;
      acc[j] = wmma_bf16(a, load_b16(bcol, hi), acc[j]);
    }
  }
#pragma unroll
  for (int jv = 0; jv < 8; ++jv) {
    int o = o0 + jv + 8 * hi;
    float bv = pbias[o];
#pragma unroll
    for (int j = 0; j < 8; ++j) {
      int n = n0 + j * 16 + lo;
      size_t idx = ((size_t)b * C_ + o) * N_ + n;
      out[idx] = x[idx] + acc[j][jv] + bv;  // residual + proj + bias
    }
  }
}

// ---------------------------------------------------------------------------
extern "C" void kernel_launch(void* const* d_in, const int* in_sizes, int n_in,
                              void* d_out, int out_size, void* d_ws,
                              size_t ws_size, hipStream_t stream) {
  const float* x      = (const float*)d_in[0];
  const float* gnw    = (const float*)d_in[1];
  const float* gnb    = (const float*)d_in[2];
  const float* qkv_w  = (const float*)d_in[3];
  const float* qkv_b  = (const float*)d_in[4];
  const float* proj_w = (const float*)d_in[5];
  const float* proj_b = (const float*)d_in[6];
  float* out = (float*)d_out;

  // workspace carve-up (bf16 tensors, 256B aligned)
  size_t off = 0;
  auto carve = [&](size_t bytes) {
    size_t o = off;
    off += (bytes + 255) & ~(size_t)255;
    return (char*)d_ws + o;
  };
  __bf16* h_t = (__bf16*)carve((size_t)B_ * N_ * C_ * 2);        // 8 MB
  __bf16* wq  = (__bf16*)carve((size_t)3 * C_ * C_ * 2);         // 1.5 MB
  __bf16* wp  = (__bf16*)carve((size_t)C_ * C_ * 2);             // 0.5 MB
  __bf16* qT  = (__bf16*)carve((size_t)B_ * HEADS * N_ * HD * 2);// 8 MB
  __bf16* kT  = (__bf16*)carve((size_t)B_ * HEADS * N_ * HD * 2);// 8 MB
  __bf16* vM  = (__bf16*)carve((size_t)B_ * HEADS * HD * N_ * 2);// 8 MB
  __bf16* o_t = (__bf16*)carve((size_t)B_ * N_ * C_ * 2);        // 8 MB

  cvt_f32_bf16<<<1024, 256, 0, stream>>>(qkv_w, wq, 3 * C_ * C_);
  cvt_f32_bf16<<<1024, 256, 0, stream>>>(proj_w, wp, C_ * C_);
  gn_kernel<<<B_ * GROUPS, 256, 0, stream>>>(x, gnw, gnb, h_t);
  qkv_gemm<<<(B_ * ((3 * C_) / 16) * (N_ / 128)) / 8, 256, 0, stream>>>(
      wq, qkv_b, h_t, qT, kT, vM);
  attn_kernel<<<B_ * HEADS * (N_ / 16), 128, 0, stream>>>(qT, kT, vM, o_t);
  proj_gemm<<<(B_ * (C_ / 16) * (N_ / 128)) / 8, 256, 0, stream>>>(
      wp, proj_b, o_t, x, out);
}
